// SelfAttention_39676907886799
// MI455X (gfx1250) — matrix operand order repaired and checked by reading, not compiled, not run on hardware
//
#include <hip/hip_runtime.h>
#include <math.h>
#include <stdint.h>

typedef __bf16 bf16_t;
typedef __attribute__((ext_vector_type(16))) __bf16 v16bf;
typedef __attribute__((ext_vector_type(4)))  __bf16 v4bf;
typedef __attribute__((ext_vector_type(8)))  float  v8f;
typedef __attribute__((ext_vector_type(4)))  unsigned int v4u;
typedef __attribute__((ext_vector_type(8)))  int    v8i;
typedef __attribute__((ext_vector_type(4)))  int    v4i;

#define WMMA_BF16(a, b, c) \
  __builtin_amdgcn_wmma_f32_16x16x32_bf16(false, (a), false, (b), (short)0, (c), false, false)

union Frag16 { v16bf v; float4 f[2]; };

// V-tile LDS layout produced by the TDM pad feature:
// 16B pad after every 1024B of data -> element offset:
//   off(r, c) = r*1040 + (c>>9)*520 + (c&511)      (bf16 elements, row = 2080B)
#define VROW 1040
#define VHALF 520

// ---------------------------------------------------------------------------
// Fragment loaders (contiguous k order: k = 16*(lane/16) + s, wave32)
// A 16x32 bf16: lanes 0-15 hold row m=lane, K {0..15}; lanes 16-31 K {16..31}
__device__ __forceinline__ v16bf lds_afrag(const bf16_t* tile, int ld) {
  const int lane = threadIdx.x & 31;
  const int row  = lane & 15;
  const int half = lane >> 4;
  const bf16_t* p = tile + row * ld + half * 16;
  Frag16 u;
  u.f[0] = *(const float4*)(p);        // K base..base+7
  u.f[1] = *(const float4*)(p + 8);    // K base+8..base+15
  return u.v;
}

// B 32x16 bf16, column n contiguous in memory (plain row-major source tile)
__device__ __forceinline__ v16bf lds_bfrag_rows(const bf16_t* tile, int ld) {
  const int lane  = threadIdx.x & 31;
  const int col   = lane & 15;
  const int khalf = lane >> 4;
  const bf16_t* p = tile + col * ld + khalf * 16;
  Frag16 u;
  u.f[0] = *(const float4*)(p);
  u.f[1] = *(const float4*)(p + 8);
  return u.v;
}

// B 32x16 bf16 from the padded V tile: B[k][n] = V[row0+n][k0+k]  (QK^T operand)
__device__ __forceinline__ v16bf lds_bfrag_vrows(const bf16_t* Vs, int row0, int k0) {
  const int lane  = threadIdx.x & 31;
  const int col   = lane & 15;
  const int khalf = lane >> 4;
  const bf16_t* p = Vs + (row0 + col) * VROW + ((k0 >> 9) * VHALF) + (k0 & 511) + khalf * 16;
  Frag16 u;
  u.f[0] = *(const float4*)(p);
  u.f[1] = *(const float4*)(p + 8);
  return u.v;
}

// B 32x16 bf16 from the padded V tile: B[k][n] = V[k0+k][dbase+n]  (PV operand, column gather)
__device__ __forceinline__ v16bf lds_bfrag_vstrided(const bf16_t* Vs, int k0, int dbase) {
  const int lane  = threadIdx.x & 31;
  const int n     = lane & 15;
  const int khalf = lane >> 4;
  const int c     = dbase + n;
  const bf16_t* p = Vs + (k0 + khalf * 16) * VROW + ((c >> 9) * VHALF) + (c & 511);
  v16bf r;
#pragma unroll
  for (int i = 0; i < 16; ++i) r[i] = p[i * VROW];
  return r;
}

// ---------------------------------------------------------------------------
__global__ void cast_f32_bf16(const float* __restrict__ in, bf16_t* __restrict__ out, int n) {
  int i = (blockIdx.x * blockDim.x + threadIdx.x) * 4;
  if (i + 3 < n) {
    float4 v = *(const float4*)(in + i);
    v4bf o;
    o[0] = (bf16_t)v.x; o[1] = (bf16_t)v.y; o[2] = (bf16_t)v.z; o[3] = (bf16_t)v.w;
    *(v4bf*)(out + i) = o;
  } else {
    for (; i < n; ++i) out[i] = (bf16_t)in[i];
  }
}

// ---------------------------------------------------------------------------
// Q = x*W1^T + b1, V = x*W2^T + b2  (shared x tile; bf16 outputs)
__global__ void __launch_bounds__(256) fc_dual(
    const bf16_t* __restrict__ xb, const bf16_t* __restrict__ w1b,
    const bf16_t* __restrict__ w2b, const float* __restrict__ b1,
    const float* __restrict__ b2, bf16_t* __restrict__ qb,
    bf16_t* __restrict__ vb, int D) {
  const int LDT = 40;
  __shared__ __align__(16) bf16_t xs[64 * 40];
  __shared__ __align__(16) bf16_t w1s[64 * 40];
  __shared__ __align__(16) bf16_t w2s[64 * 40];

  const int m0   = blockIdx.y * 64;
  const int n0   = blockIdx.x * 64;
  const int tid  = threadIdx.x;
  const int wave = tid >> 5;
  const int lane = tid & 31;
  const int mt   = wave >> 1;
  const int ntb  = (wave & 1) * 2;

  v8f accQ[2], accV[2];
#pragma unroll
  for (int j = 0; j < 2; ++j)
#pragma unroll
    for (int r = 0; r < 8; ++r) { accQ[j][r] = 0.f; accV[j][r] = 0.f; }

  const int r = tid >> 2;
  const int c = (tid & 3) * 8;
  for (int k0 = 0; k0 < D; k0 += 32) {
    __syncthreads();
    *(float4*)(xs  + r * LDT + c) = *(const float4*)(xb  + (size_t)(m0 + r) * D + k0 + c);
    *(float4*)(w1s + r * LDT + c) = *(const float4*)(w1b + (size_t)(n0 + r) * D + k0 + c);
    *(float4*)(w2s + r * LDT + c) = *(const float4*)(w2b + (size_t)(n0 + r) * D + k0 + c);
    __syncthreads();
    v16bf a = lds_afrag(xs + mt * 16 * LDT, LDT);
#pragma unroll
    for (int j = 0; j < 2; ++j) {
      v16bf bq = lds_bfrag_rows(w1s + (ntb + j) * 16 * LDT, LDT);
      accQ[j] = WMMA_BF16(a, bq, accQ[j]);
      v16bf bv = lds_bfrag_rows(w2s + (ntb + j) * 16 * LDT, LDT);
      accV[j] = WMMA_BF16(a, bv, accV[j]);
    }
  }

  const int nin = lane & 15;
  const int mo  = (lane >> 4) * 8;
#pragma unroll
  for (int j = 0; j < 2; ++j) {
    const int e = n0 + (ntb + j) * 16 + nin;
    const float biasQ = b1[e];
    const float biasV = b2[e];
#pragma unroll
    for (int rr = 0; rr < 8; ++rr) {
      const int m = m0 + mt * 16 + mo + rr;
      qb[(size_t)m * D + e] = (bf16_t)(accQ[j][rr] + biasQ);
      vb[(size_t)m * D + e] = (bf16_t)(accV[j][rr] + biasV);
    }
  }
}

// ---------------------------------------------------------------------------
// Fused flash attention. V tiles are streamed into LDS by the Tensor Data
// Mover (tensor_load_to_lds, TENSORcnt) with the pad feature supplying the
// bank-conflict pad (16B per 1024B of data).
__global__ void __launch_bounds__(256) flash_attn(
    const bf16_t* __restrict__ qb, const bf16_t* __restrict__ vb,
    float* __restrict__ out, int B, int L, int D) {
  extern __shared__ char smem[];
  const int LDQ  = D + 8;   // 1032 (Q tile keeps the simple pad; loaded once by VALU)
  const int LDSS = 72;

  bf16_t* Vs   = (bf16_t*)smem;                 // 64 rows x VROW bf16 (TDM target, first!)
  bf16_t* Qs   = Vs + 64 * VROW;                // 32 x LDQ bf16
  float*  Ss   = (float*)(Qs + 32 * LDQ);       // 32 x LDSS f32
  bf16_t* Ps   = (bf16_t*)(Ss + 32 * LDSS);     // 32 x LDSS bf16
  float*  rowm = (float*)(Ps + 32 * LDSS);      // 32
  float*  rowl = rowm + 32;                     // 32
  float*  rsc  = rowl + 32;                     // 32

  const int b    = blockIdx.y;
  const int q0   = blockIdx.x * 32;
  const int tid  = threadIdx.x;
  const int wave = tid >> 5;
  const int lane = tid & 31;

  const bf16_t* Qg = qb + ((size_t)b * L + q0) * D;
  const bf16_t* Vg = vb + (size_t)b * L * D;

  const int vecsPerRow = D / 8;
  for (int idx = tid; idx < 32 * vecsPerRow; idx += 256) {
    const int rr = idx / vecsPerRow;
    const int cc = (idx % vecsPerRow) * 8;
    *(float4*)(Qs + rr * LDQ + cc) = *(const float4*)(Qg + (size_t)rr * D + cc);
  }
  if (tid < 32) { rowm[tid] = -__builtin_inff(); rowl[tid] = 0.f; }

  v8f o[2][8];
#pragma unroll
  for (int mt = 0; mt < 2; ++mt)
#pragma unroll
    for (int nt = 0; nt < 8; ++nt)
#pragma unroll
      for (int rr = 0; rr < 8; ++rr) o[mt][nt][rr] = 0.f;

  const int smt = wave >> 2;
  const int snt = wave & 3;
  const int dbase = wave * 128;
  const int mo = (lane >> 4) * 8;
  const int nin = lane & 15;

  // Constant TDM descriptor group 1:
  // data_size=3 (8B units) | pad_enable | pad_interval=7 (1024B) | pad_amount=3 (16B)
  // tensor_dim0 = tile_dim0 = tensor_dim0_stride = 256 (8B units = 2048B row = D bf16)
  // tensor_dim1 = tile_dim1 = 64 rows
  v8i tg1;
  tg1[0] = (int)((3u << 16) | (1u << 20) | (7u << 22) | (3u << 25));
  tg1[1] = (int)(256u << 16);   // [63:48] tensor_dim0 lo
  tg1[2] = (int)(64u << 16);    // [95:80] tensor_dim1 lo
  tg1[3] = (int)(256u << 16);   // [127:112] tile_dim0
  tg1[4] = 64;                  // [143:128] tile_dim1
  tg1[5] = 256;                 // [191:160] tensor_dim0_stride lo
  tg1[6] = 0;
  tg1[7] = 0;
  v4i z4; z4[0] = 0; z4[1] = 0; z4[2] = 0; z4[3] = 0;
  v8i z8; z8[0] = 0; z8[1] = 0; z8[2] = 0; z8[3] = 0;
  z8[4] = 0; z8[5] = 0; z8[6] = 0; z8[7] = 0;
  const unsigned int vs_lds = (unsigned int)(uintptr_t)Vs;

  for (int kb = 0; kb < L; kb += 64) {
    __syncthreads();  // everyone done with the previous Vs tile
    const bf16_t* Vt = Vg + (size_t)kb * D;
    if (wave == 0) {
      const uint64_t ga = (uint64_t)(uintptr_t)Vt;
      v4u g0;
      g0[0] = 1u;                                   // count=1, user descriptor
      g0[1] = vs_lds;                               // lds_addr (bytes)
      g0[2] = (unsigned int)ga;                     // global_addr lo (D# bits 95:64)
      g0[3] = (unsigned int)((ga >> 32) & 0x01FFFFFFu) | 0x80000000u;  // gaddr hi | type=2
      __builtin_amdgcn_tensor_load_to_lds(g0, tg1, z4, z4, z8, 0);
      __builtin_amdgcn_s_wait_tensorcnt(0);
    }
    __syncthreads();

    // Phase 1: S(32x64) = Qblk * Vtile^T ; each wave one 16x16 tile, K = D
    v8f s;
#pragma unroll
    for (int rr = 0; rr < 8; ++rr) s[rr] = 0.f;
    for (int k0 = 0; k0 < D; k0 += 32) {
      v16bf a  = lds_afrag(Qs + smt * 16 * LDQ + k0, LDQ);
      v16bf bk = lds_bfrag_vrows(Vs, snt * 16, k0);
      s = WMMA_BF16(a, bk, s);
    }
#pragma unroll
    for (int rr = 0; rr < 8; ++rr)
      Ss[(smt * 16 + mo + rr) * LDSS + snt * 16 + nin] = s[rr];
    __syncthreads();

    // Phase 2: online softmax (one thread per query row)
    if (tid < 32) {
      const float mprev = rowm[tid];
      float mmax = mprev;
      const float* srow = Ss + tid * LDSS;
      for (int j = 0; j < 64; ++j) mmax = fmaxf(mmax, srow[j]);
      const float scale = __expf(mprev - mmax);
      float sum = 0.f;
      bf16_t* prow = Ps + tid * LDSS;
      for (int j = 0; j < 64; ++j) {
        const float p = __expf(srow[j] - mmax);
        sum += p;
        prow[j] = (bf16_t)p;
      }
      rowm[tid] = mmax;
      rowl[tid] = rowl[tid] * scale + sum;
      rsc[tid]  = scale;
    }
    __syncthreads();

    // Phase 3: O = O*scale + P(32x64) * Vtile(64 x my 128 cols)
#pragma unroll
    for (int mt = 0; mt < 2; ++mt)
#pragma unroll
      for (int rr = 0; rr < 8; ++rr) {
        const float sc = rsc[mt * 16 + mo + rr];
#pragma unroll
        for (int nt = 0; nt < 8; ++nt) o[mt][nt][rr] *= sc;
      }
#pragma unroll
    for (int ks = 0; ks < 2; ++ks) {
      v16bf a0 = lds_afrag(Ps + 0 * 16 * LDSS + ks * 32, LDSS);
      v16bf a1 = lds_afrag(Ps + 1 * 16 * LDSS + ks * 32, LDSS);
#pragma unroll
      for (int nt = 0; nt < 8; ++nt) {
        v16bf bv = lds_bfrag_vstrided(Vs, ks * 32, dbase + nt * 16);
        o[0][nt] = WMMA_BF16(a0, bv, o[0][nt]);
        o[1][nt] = WMMA_BF16(a1, bv, o[1][nt]);
      }
    }
  }

  __syncthreads();
#pragma unroll
  for (int mt = 0; mt < 2; ++mt)
#pragma unroll
    for (int rr = 0; rr < 8; ++rr) {
      const int m = mt * 16 + mo + rr;
      const float inv = 1.0f / rowl[m];
      float* orow = out + ((size_t)b * L + q0 + m) * D + dbase;
#pragma unroll
      for (int nt = 0; nt < 8; ++nt) orow[nt * 16 + nin] = o[mt][nt][rr] * inv;
    }
}

// ---------------------------------------------------------------------------
extern "C" void kernel_launch(void* const* d_in, const int* in_sizes, int n_in,
                              void* d_out, int out_size, void* d_ws, size_t ws_size,
                              hipStream_t stream) {
  const int B = 4, L = 2048, D = 1024;
  const int M = B * L;

  const float* x  = (const float*)d_in[0];
  const float* W1 = (const float*)d_in[1];
  const float* b1 = (const float*)d_in[2];
  const float* W2 = (const float*)d_in[3];
  const float* b2 = (const float*)d_in[4];
  float* out = (float*)d_out;

  char* ws = (char*)d_ws;
  bf16_t* xb  = (bf16_t*)ws;               ws += (size_t)M * D * sizeof(bf16_t);
  bf16_t* w1b = (bf16_t*)ws;               ws += (size_t)D * D * sizeof(bf16_t);
  bf16_t* w2b = (bf16_t*)ws;               ws += (size_t)D * D * sizeof(bf16_t);
  bf16_t* qb  = (bf16_t*)ws;               ws += (size_t)M * D * sizeof(bf16_t);
  bf16_t* vb  = (bf16_t*)ws;               ws += (size_t)M * D * sizeof(bf16_t);

  {
    const int nx = M * D;
    cast_f32_bf16<<<dim3((nx / 4 + 255) / 256), 256, 0, stream>>>(x, xb, nx);
    const int nw = D * D;
    cast_f32_bf16<<<dim3((nw / 4 + 255) / 256), 256, 0, stream>>>(W1, w1b, nw);
    cast_f32_bf16<<<dim3((nw / 4 + 255) / 256), 256, 0, stream>>>(W2, w2b, nw);
  }

  fc_dual<<<dim3(D / 64, M / 64), 256, 0, stream>>>(xb, w1b, w2b, b1, b2, qb, vb, D);

  {
    const int LDQ = D + 8;
    const size_t smem = (size_t)64 * VROW * 2 +        // padded V tile (TDM target)
                        (size_t)32 * LDQ * 2 +
                        (size_t)32 * 72 * 4 + (size_t)32 * 72 * 2 + 3 * 32 * 4;
    (void)hipFuncSetAttribute((const void*)flash_attn,
                              hipFuncAttributeMaxDynamicSharedMemorySize, (int)smem);
    flash_attn<<<dim3(L / 32, B), 256, smem, stream>>>(qb, vb, out, B, L, D);
  }
}
